// TENProLayer_25082609008798
// MI455X (gfx1250) — matrix operand order repaired and checked by reading, not compiled, not run on hardware
//
#include <hip/hip_runtime.h>

// ---------------------------------------------------------------------------
// Problem constants (from reference): B=4, T=4096, D=1024, K=128, H=8, KH=16,
// KER=4, M=4*D=4096.  Ntok = B*T = 16384.
// ---------------------------------------------------------------------------
#define BB   4
#define TT   4096
#define DD   1024
#define KK   128
#define K2   256
#define MM   4096
#define NTOK (BB*TT)

#define DEV __device__ __forceinline__

typedef __attribute__((ext_vector_type(16))) __bf16         v16bf;
typedef __attribute__((ext_vector_type(8)))  float          v8f;
typedef __attribute__((ext_vector_type(8)))  unsigned short v8us;

union FragBF {
  struct { v8us lo; v8us hi; } p;
  v16bf b;
};

DEV unsigned short f2bf(float f) {
  unsigned int u = __builtin_bit_cast(unsigned int, f);
  u += 0x7FFFu + ((u >> 16) & 1u);      // round-to-nearest-even
  return (unsigned short)(u >> 16);
}
DEV float sigmoidf(float x) { return 1.0f / (1.0f + __expf(-x)); }
DEV float siluf(float x)    { return x * sigmoidf(x); }

// ---------------------------------------------------------------------------
// fp32 -> bf16 conversion
// ---------------------------------------------------------------------------
__global__ void k_cvt_bf16(const float* __restrict__ in,
                           unsigned short* __restrict__ out, int n) {
  int i = blockIdx.x * blockDim.x + threadIdx.x;
  if (i < n) out[i] = f2bf(in[i]);
}

// ---------------------------------------------------------------------------
// LayerNorm (one block per token row), writes bf16 normalized output
// ---------------------------------------------------------------------------
__global__ void k_layernorm(const float* __restrict__ x,
                            const float* __restrict__ g,
                            const float* __restrict__ b,
                            unsigned short* __restrict__ outbf, int D) {
  __shared__ float s1[256], s2[256];
  const int row = blockIdx.x, tid = threadIdx.x;
  const float* xr = x + (size_t)row * D;
  float s = 0.f, q = 0.f;
  for (int i = tid; i < D; i += 256) { float v = xr[i]; s += v; q += v * v; }
  s1[tid] = s; s2[tid] = q; __syncthreads();
  for (int o = 128; o > 0; o >>= 1) {
    if (tid < o) { s1[tid] += s1[tid + o]; s2[tid] += s2[tid + o]; }
    __syncthreads();
  }
  const float mean = s1[0] / (float)D;
  const float var  = s2[0] / (float)D - mean * mean;
  const float rstd = rsqrtf(var + 1e-5f);
  for (int i = tid; i < D; i += 256) {
    float v = (xr[i] - mean) * rstd * g[i] + b[i];
    outbf[(size_t)row * D + i] = f2bf(v);
  }
}

// ---------------------------------------------------------------------------
// bf16 WMMA GEMM:  C[M,N] = A[M,Kd] @ W[N,Kd]^T   (fp32 accumulate)
// block = 256 thr = 8 waves (4 row-waves x 2 col-waves); each wave owns a
// 32x64 output region = 2x4 grid of 16x16 WMMA tiles -> 8 WMMAs per 32-K
// step against 12 b128 loads (2.67x operand reuse).  Block tile 128x128.
// __launch_bounds__(256,1): one block per EU so the register allocator can
// keep the whole pipelined schedule (8 v8f accs + double-buffered fragments)
// in VGPRs without scratch spills.
// Requires M%128==0, N%128==0, Kd%32==0.
// EPI: 0 = plain fp32 store; 1 = silu(acc+bias) -> bf16 store;
//      2 = acc + bias + add -> fp32 store (fused residual)
// ---------------------------------------------------------------------------
template <int EPI>
__global__ __launch_bounds__(256, 1)
void k_gemm_bf16(const unsigned short* __restrict__ A,
                 const unsigned short* __restrict__ W,
                 const float* __restrict__ bias,
                 const float* __restrict__ add,
                 float* __restrict__ Cf,
                 unsigned short* __restrict__ Cb,
                 int Mrows, int N, int Kd) {
  const int lane    = threadIdx.x & 31;
  const int wave    = threadIdx.x >> 5;   // 0..7
  const int waveRow = wave & 3;           // 4 row-waves
  const int waveCol = wave >> 2;          // 2 col-waves
  const int rowBase = blockIdx.y * 128 + waveRow * 32;
  const int colBase = blockIdx.x * 128 + waveCol * 64;
  const int rIn     = lane & 15;
  const int hi      = lane >> 4;          // lane half
  const int kA      = hi * 8;             // A: K halves {0..7,16..23}/{8..15,24..31}
  const int kB      = hi * 8;             // B: K halves {0..7,16..23}/{8..15,24..31}

  v8f acc[2][4];
#pragma unroll
  for (int i = 0; i < 2; i++)
#pragma unroll
    for (int j = 0; j < 4; j++)
      acc[i][j] = (v8f){0.f, 0.f, 0.f, 0.f, 0.f, 0.f, 0.f, 0.f};

  size_t rA[2], cB[4];
#pragma unroll
  for (int i = 0; i < 2; i++) rA[i] = (size_t)(rowBase + i * 16 + rIn) * (size_t)Kd;
#pragma unroll
  for (int j = 0; j < 4; j++) cB[j] = (size_t)(colBase + j * 16 + rIn) * (size_t)Kd;

  for (int k0 = 0; k0 < Kd; k0 += 32) {
    FragBF a[2], b[4];
#pragma unroll
    for (int i = 0; i < 2; i++) {
      a[i].p.lo = *(const v8us*)(A + rA[i] + k0 + kA);
      a[i].p.hi = *(const v8us*)(A + rA[i] + k0 + 16 + kA);
    }
#pragma unroll
    for (int j = 0; j < 4; j++) {
      b[j].p.lo = *(const v8us*)(W + cB[j] + k0 + kB);
      b[j].p.hi = *(const v8us*)(W + cB[j] + k0 + 16 + kB);
    }
    // prefetch the streaming A rows a few K-steps ahead (global_prefetch_b8)
    if (k0 + 256 < Kd) {
      __builtin_prefetch(A + rA[0] + k0 + 256, 0, 1);
      __builtin_prefetch(A + rA[1] + k0 + 256, 0, 1);
    }
#pragma unroll
    for (int i = 0; i < 2; i++)
#pragma unroll
      for (int j = 0; j < 4; j++)
        acc[i][j] = __builtin_amdgcn_wmma_f32_16x16x32_bf16(
            false, a[i].b, false, b[j].b, (short)0, acc[i][j], false, false);
  }

#pragma unroll
  for (int i = 0; i < 2; i++)
#pragma unroll
    for (int j = 0; j < 4; j++) {
#pragma unroll
      for (int e = 0; e < 8; e++) {
        const int row = rowBase + i * 16 + e + hi * 8;
        const int col = colBase + j * 16 + (lane & 15);
        const size_t idx = (size_t)row * (size_t)N + col;
        float v = acc[i][j][e];
        if (EPI == 0) {
          Cf[idx] = v;
        } else if (EPI == 1) {
          v += bias[col];
          Cb[idx] = f2bf(siluf(v));
        } else {
          v += bias[col] + add[idx];
          Cf[idx] = v;
        }
      }
    }
}

// ---------------------------------------------------------------------------
// Depthwise causal conv (KER=4) + bias + SiLU + gated memory injection.
// Token-major layout [Ntok, 256]; cols 0..127 real, 128..255 imag.
// ---------------------------------------------------------------------------
__global__ void k_conv_mem(const float* __restrict__ beta_lin,
                           const float* __restrict__ conv_w,
                           const float* __restrict__ conv_b,
                           const float* __restrict__ memout,
                           const float* __restrict__ mem_gate,
                           float* __restrict__ brbi) {
  const size_t idx = (size_t)blockIdx.x * blockDim.x + threadIdx.x;
  const size_t total = (size_t)NTOK * K2;
  if (idx >= total) return;
  const int ch = (int)(idx & (K2 - 1));
  const size_t tok = idx >> 8;
  const int t = (int)(tok % TT);
  float acc = 0.f;
#pragma unroll
  for (int q = 0; q < 4; q++) {
    const int tt = t - 3 + q;
    if (tt >= 0) acc += conv_w[ch * 4 + q] * beta_lin[idx - (size_t)(3 - q) * K2];
  }
  const float mg = sigmoidf(mem_gate[0]);
  brbi[idx] = siluf(acc + conv_b[ch]) + mg * memout[idx];
}

// ---------------------------------------------------------------------------
// Complex linear recurrence c[t] = lam*c[t-1] + beta[t] as a chunked scan.
// One block per (b,k): 256 threads x 16 steps, Hillis-Steele combine in LDS.
// (Mathematically identical to the reference's zero-padded FFT convolution.)
// ---------------------------------------------------------------------------
__global__ void k_scan(const float* __restrict__ brbi,
                       const float* __restrict__ log_decay,
                       const float* __restrict__ freq,
                       float* __restrict__ cbuf) {
  __shared__ float sar[256], sai[256], sbr[256], sbi[256];
  const int b   = blockIdx.x >> 7;
  const int k   = blockIdx.x & 127;
  const int tid = threadIdx.x;
  const float mag = sigmoidf(log_decay[k]);
  const float lr  = mag * __cosf(freq[k]);
  const float li  = mag * __sinf(freq[k]);
  float ar = lr, ai = li;                        // lam^16 by squaring
#pragma unroll
  for (int s = 0; s < 4; s++) { float nr = ar * ar - ai * ai, ni = 2.f * ar * ai; ar = nr; ai = ni; }

  const int CH = TT / 256;                       // 16
  const size_t base = ((size_t)b * TT) * K2 + k;
  const int t0 = tid * CH;

  float vr = 0.f, vi = 0.f;                      // pass 1: chunk-local value
  for (int t = t0; t < t0 + CH; t++) {
    const float br = brbi[base + (size_t)t * K2];
    const float bi = brbi[base + (size_t)t * K2 + KK];
    const float nr = lr * vr - li * vi + br;
    const float ni = lr * vi + li * vr + bi;
    vr = nr; vi = ni;
  }
  sar[tid] = ar; sai[tid] = ai; sbr[tid] = vr; sbi[tid] = vi;
  __syncthreads();
  for (int off = 1; off < 256; off <<= 1) {
    float par = 0.f, pai = 0.f, pbr = 0.f, pbi = 0.f;
    if (tid >= off) { par = sar[tid - off]; pai = sai[tid - off];
                      pbr = sbr[tid - off]; pbi = sbi[tid - off]; }
    __syncthreads();
    if (tid >= off) {
      const float car = sar[tid], cai = sai[tid], cbr = sbr[tid], cbi = sbi[tid];
      sar[tid] = car * par - cai * pai;          // a = a_cur * a_prev
      sai[tid] = car * pai + cai * par;
      sbr[tid] = car * pbr - cai * pbi + cbr;    // b = a_cur * b_prev + b_cur
      sbi[tid] = car * pbi + cai * pbr + cbi;
    }
    __syncthreads();
  }
  float cr0 = 0.f, ci0 = 0.f;
  if (tid > 0) { cr0 = sbr[tid - 1]; ci0 = sbi[tid - 1]; }

  vr = cr0; vi = ci0;                            // pass 2: emit with carry-in
  for (int t = t0; t < t0 + CH; t++) {
    const float br = brbi[base + (size_t)t * K2];
    const float bi = brbi[base + (size_t)t * K2 + KK];
    const float nr = lr * vr - li * vi + br;
    const float ni = lr * vi + li * vr + bi;
    vr = nr; vi = ni;
    cbuf[base + (size_t)t * K2]      = vr;
    cbuf[base + (size_t)t * K2 + KK] = vi;
  }
}

// ---------------------------------------------------------------------------
// Per-head coupling (16x16 per head) applied to real & imag parts.
// One block per token; coupling cached in LDS. Writes fp32 eig (output) and
// a bf16 copy (GEMM input).
// ---------------------------------------------------------------------------
__global__ void k_coupling(const float* __restrict__ cbuf,
                           const float* __restrict__ coupling,
                           float* __restrict__ eig_f,
                           unsigned short* __restrict__ eig_bf) {
  __shared__ float sc[8 * 16 * 16];
  const int tid = threadIdx.x;
  for (int i = tid; i < 2048; i += 256) sc[i] = coupling[i];
  __syncthreads();
  const size_t tok = blockIdx.x;
  const int ch = tid;
  const int im = ch >> 7;
  const int hk = ch & 127;
  const int h  = hk >> 4;
  const int j  = hk & 15;
  const float* crow = cbuf + tok * K2 + im * KK + h * 16;
  const float* w    = sc + (h * 16 + j) * 16;
  float acc = 0.f;
#pragma unroll
  for (int kk = 0; kk < 16; kk++) acc += w[kk] * crow[kk];
  eig_f[tok * K2 + ch]  = acc;
  eig_bf[tok * K2 + ch] = f2bf(acc);
}

// ---------------------------------------------------------------------------
// x1 = x + sigmoid(gate_pre + gate_b) * op   (op overwritten in-place by x1)
// ---------------------------------------------------------------------------
__global__ void k_gate_res(const float* __restrict__ x,
                           const float* __restrict__ gate_pre,
                           const float* __restrict__ gate_b,
                           float* __restrict__ x1_op) {
  const size_t idx = (size_t)blockIdx.x * blockDim.x + threadIdx.x;
  const size_t total = (size_t)NTOK * DD;
  if (idx >= total) return;
  const int col = (int)(idx & (DD - 1));
  const float g = sigmoidf(gate_pre[idx] + gate_b[col]);
  x1_op[idx] = x[idx] + g * x1_op[idx];
}

// ---------------------------------------------------------------------------
// Host-side orchestration
// ---------------------------------------------------------------------------
static inline size_t align_up(size_t v) { return (v + 255) & ~(size_t)255; }

extern "C" void kernel_launch(void* const* d_in, const int* in_sizes, int n_in,
                              void* d_out, int out_size, void* d_ws, size_t ws_size,
                              hipStream_t stream) {
  (void)in_sizes; (void)n_in; (void)out_size; (void)ws_size;

  const float* x         = (const float*)d_in[0];
  const float* prev_eig  = (const float*)d_in[1];
  const float* in_proj_w = (const float*)d_in[2];
  const float* conv_w    = (const float*)d_in[3];
  const float* conv_b    = (const float*)d_in[4];
  const float* mem_gate  = (const float*)d_in[5];
  const float* mem_proj_w= (const float*)d_in[6];
  const float* log_decay = (const float*)d_in[7];
  const float* frequency = (const float*)d_in[8];
  const float* coupling  = (const float*)d_in[9];
  const float* out_proj_w= (const float*)d_in[10];
  const float* gate_w    = (const float*)d_in[11];
  const float* gate_b    = (const float*)d_in[12];
  const float* mlp_w1    = (const float*)d_in[13];
  const float* mlp_b1    = (const float*)d_in[14];
  const float* mlp_w2    = (const float*)d_in[15];
  const float* mlp_b2    = (const float*)d_in[16];
  const float* n1_g      = (const float*)d_in[17];
  const float* n1_b      = (const float*)d_in[18];
  const float* n2_g      = (const float*)d_in[19];
  const float* n2_b      = (const float*)d_in[20];

  float* out_x2  = (float*)d_out;                       // [Ntok, D]
  float* out_eig = (float*)d_out + (size_t)NTOK * DD;   // [Ntok, 2K]

  // ---- workspace layout (bytes) ----
  char* ws = (char*)d_ws;
  size_t ofs = 0;
  auto alloc = [&](size_t bytes) { size_t o = ofs; ofs = align_up(ofs + bytes); return o; };
  const size_t o_w_in   = alloc((size_t)K2 * DD * 2);
  const size_t o_w_mem  = alloc((size_t)K2 * K2 * 2);
  const size_t o_w_gate = alloc((size_t)DD * DD * 2);
  const size_t o_w_out  = alloc((size_t)DD * K2 * 2);
  const size_t o_w1     = alloc((size_t)MM * DD * 2);
  const size_t o_w2     = alloc((size_t)DD * MM * 2);
  const size_t o_xn     = alloc((size_t)NTOK * DD * 2);
  const size_t o_prev   = alloc((size_t)NTOK * K2 * 2);
  const size_t o_beta   = alloc((size_t)NTOK * K2 * 4);
  const size_t o_memout = alloc((size_t)NTOK * K2 * 4);
  const size_t o_brbi   = alloc((size_t)NTOK * K2 * 4);
  const size_t o_cbuf   = alloc((size_t)NTOK * K2 * 4);
  const size_t o_eigbf  = alloc((size_t)NTOK * K2 * 2);
  const size_t o_gpre   = alloc((size_t)NTOK * DD * 4);
  const size_t o_x1     = alloc((size_t)NTOK * DD * 4);   // first op, then x1 in-place
  const size_t o_ybf    = alloc((size_t)NTOK * DD * 2);
  const size_t o_hid    = alloc((size_t)NTOK * MM * 2);

  unsigned short* w_in_bf   = (unsigned short*)(ws + o_w_in);
  unsigned short* w_mem_bf  = (unsigned short*)(ws + o_w_mem);
  unsigned short* w_gate_bf = (unsigned short*)(ws + o_w_gate);
  unsigned short* w_out_bf  = (unsigned short*)(ws + o_w_out);
  unsigned short* w1_bf     = (unsigned short*)(ws + o_w1);
  unsigned short* w2_bf     = (unsigned short*)(ws + o_w2);
  unsigned short* xn_bf     = (unsigned short*)(ws + o_xn);
  unsigned short* prev_bf   = (unsigned short*)(ws + o_prev);
  float*          beta_lin  = (float*)(ws + o_beta);
  float*          memout    = (float*)(ws + o_memout);
  float*          brbi      = (float*)(ws + o_brbi);
  float*          cbuf      = (float*)(ws + o_cbuf);
  unsigned short* eig_bf    = (unsigned short*)(ws + o_eigbf);
  float*          gate_pre  = (float*)(ws + o_gpre);
  float*          x1        = (float*)(ws + o_x1);
  unsigned short* y_bf      = (unsigned short*)(ws + o_ybf);
  unsigned short* hid_bf    = (unsigned short*)(ws + o_hid);

  auto cvt = [&](const float* src, unsigned short* dst, int n) {
    k_cvt_bf16<<<(n + 255) / 256, 256, 0, stream>>>(src, dst, n);
  };

  // 1) weight / prev_eig conversion to bf16
  cvt(in_proj_w,  w_in_bf,   K2 * DD);
  cvt(mem_proj_w, w_mem_bf,  K2 * K2);
  cvt(gate_w,     w_gate_bf, DD * DD);
  cvt(out_proj_w, w_out_bf,  DD * K2);
  cvt(mlp_w1,     w1_bf,     MM * DD);
  cvt(mlp_w2,     w2_bf,     DD * MM);
  cvt(prev_eig,   prev_bf,   NTOK * K2);

  // 2) LN1 -> xn (bf16)
  k_layernorm<<<NTOK, 256, 0, stream>>>(x, n1_g, n1_b, xn_bf, DD);

  // 3) GEMMs: beta_lin = xn @ in_proj^T ; memout = prev @ mem_proj^T ; gate_pre
  {
    dim3 g(K2 / 128, NTOK / 128);
    k_gemm_bf16<0><<<g, 256, 0, stream>>>(xn_bf, w_in_bf, nullptr, nullptr,
                                          beta_lin, nullptr, NTOK, K2, DD);
  }
  {
    dim3 g(K2 / 128, NTOK / 128);
    k_gemm_bf16<0><<<g, 256, 0, stream>>>(prev_bf, w_mem_bf, nullptr, nullptr,
                                          memout, nullptr, NTOK, K2, K2);
  }
  {
    dim3 g(DD / 128, NTOK / 128);
    k_gemm_bf16<0><<<g, 256, 0, stream>>>(xn_bf, w_gate_bf, nullptr, nullptr,
                                          gate_pre, nullptr, NTOK, DD, DD);
  }

  // 4) conv + SiLU + memory injection
  k_conv_mem<<<(NTOK * K2) / 256, 256, 0, stream>>>(beta_lin, conv_w, conv_b,
                                                    memout, mem_gate, brbi);

  // 5) complex recurrence scan
  k_scan<<<BB * KK, 256, 0, stream>>>(brbi, log_decay, frequency, cbuf);

  // 6) per-head coupling -> eig (fp32 output + bf16 copy)
  k_coupling<<<NTOK, 256, 0, stream>>>(cbuf, coupling, out_eig, eig_bf);

  // 7) op = eig @ out_proj^T ; then gated residual x1 = x + sig(gate)*op
  {
    dim3 g(DD / 128, NTOK / 128);
    k_gemm_bf16<0><<<g, 256, 0, stream>>>(eig_bf, w_out_bf, nullptr, nullptr,
                                          x1, nullptr, NTOK, DD, K2);
  }
  k_gate_res<<<(NTOK * DD) / 256, 256, 0, stream>>>(x, gate_pre, gate_b, x1);

  // 8) LN2 -> y (bf16); MLP: hidden = silu(y@w1^T+b1) (bf16);
  //    x2 = x1 + hidden@w2^T + b2  (written straight to d_out)
  k_layernorm<<<NTOK, 256, 0, stream>>>(x1, n2_g, n2_b, y_bf, DD);
  {
    dim3 g(MM / 128, NTOK / 128);
    k_gemm_bf16<1><<<g, 256, 0, stream>>>(y_bf, w1_bf, mlp_b1, nullptr,
                                          nullptr, hid_bf, NTOK, MM, DD);
  }
  {
    dim3 g(DD / 128, NTOK / 128);
    k_gemm_bf16<2><<<g, 256, 0, stream>>>(hid_bf, w2_bf, mlp_b2, x1,
                                          out_x2, nullptr, NTOK, DD, MM);
  }
}
